// AUGRU_32049045963131
// MI455X (gfx1250) — matrix operand order repaired and checked by reading, not compiled, not run on hardware
//
#include <hip/hip_runtime.h>
#include <hip/hip_bf16.h>

#define BATCH 32768
#define DK 512
#define DN 512

typedef __attribute__((ext_vector_type(16))) __bf16 v16bf;
typedef __attribute__((ext_vector_type(8)))  float  v8f;

union Frag {
    uint4 q[2];
    v16bf v;
};

__device__ __forceinline__ unsigned int f2bf1(float f) {
    unsigned int u = __float_as_uint(f);
    return (u + 0x7FFFu + ((u >> 16) & 1u)) >> 16;   // RNE f32 -> bf16
}

__device__ __forceinline__ float sigmoidf_(float x) {
    return 1.0f / (1.0f + __expf(-x));
}
__device__ __forceinline__ float tanhf_(float x) {
    return 1.0f - 2.0f / (__expf(2.0f * x) + 1.0f);
}

// ---- pre-pass: f32 -> bf16, 8 elements per thread -----------------------
__global__ void cvt_bf16x8(const float* __restrict__ s,
                           unsigned short* __restrict__ d, int n) {
    int i = (blockIdx.x * blockDim.x + threadIdx.x) * 8;
    if (i >= n) return;
    const float4* p = (const float4*)(s + i);
    float4 f0 = p[0], f1 = p[1];
    uint4 o;
    o.x = f2bf1(f0.x) | (f2bf1(f0.y) << 16);
    o.y = f2bf1(f0.z) | (f2bf1(f0.w) << 16);
    o.z = f2bf1(f1.x) | (f2bf1(f1.y) << 16);
    o.w = f2bf1(f1.z) | (f2bf1(f1.w) << 16);
    *(uint4*)(d + i) = o;
}

// ---- pre-pass: weight convert + transpose ([K][N] f32 -> [g][N][K] bf16) -
__global__ void wtrans(const float* __restrict__ W0, const float* __restrict__ W1,
                       const float* __restrict__ W2, const float* __restrict__ W3,
                       const float* __restrict__ W4, const float* __restrict__ W5,
                       unsigned short* __restrict__ WxT,
                       unsigned short* __restrict__ WhT) {
    __shared__ unsigned short tile[32][33];
    const int g = blockIdx.z;
    const float* W = (g == 0) ? W0 : (g == 1) ? W1 : (g == 2) ? W2
                   : (g == 3) ? W3 : (g == 4) ? W4 : W5;
    unsigned short* out = (g < 3) ? (WxT + (size_t)g * DN * DK)
                                  : (WhT + (size_t)(g - 3) * DN * DK);
    const int kb = blockIdx.x * 32, nb = blockIdx.y * 32;
    const int tx = threadIdx.x, ty = threadIdx.y;     // 32 x 8
#pragma unroll
    for (int i = 0; i < 32; i += 8)
        tile[ty + i][tx] = (unsigned short)f2bf1(W[(size_t)(kb + ty + i) * DN + nb + tx]);
    __syncthreads();
#pragma unroll
    for (int i = 0; i < 32; i += 8)
        out[(size_t)(nb + ty + i) * DK + kb + tx] = tile[tx][ty + i];
}

// ---- fused AUGRU: 6 GEMMs via bf16 WMMA + gate epilogue ------------------
// wave tile: 32 rows x 32 cols (2 M-tiles x 2 N-tiles).
// Block: 8 waves -> 256 rows x 32 cols. Every B fragment feeds 2 WMMAs.
#define LOADB(dst, base, off) { const uint4* p_ = (const uint4*)((base) + (off)); \
                                (dst).q[0] = p_[0]; (dst).q[1] = p_[1]; }
#define LOADA(dst, base, off) { const uint4* p_ = (const uint4*)((base) + (off)); \
                                (dst).q[0] = p_[0]; (dst).q[1] = p_[1]; }
#define MMA(acc, A, Bf) acc = __builtin_amdgcn_wmma_f32_16x16x32_bf16( \
        false, (A).v, false, (Bf).v, (short)0, acc, false, false)

__global__ __launch_bounds__(256, 1) void augru_wmma(
    const unsigned short* __restrict__ xb,   // [B][DK] bf16
    const unsigned short* __restrict__ hb,   // [B][DK] bf16
    const unsigned short* __restrict__ WxT,  // [3][DN][DK] bf16 (u,r,c)
    const unsigned short* __restrict__ WhT,  // [3][DN][DK] bf16 (u,r,c)
    const float* __restrict__ bu, const float* __restrict__ br,
    const float* __restrict__ bc, const float* __restrict__ att,
    const float* __restrict__ state, float* __restrict__ out) {
    const int lane = threadIdx.x & 31;
    const int wave = threadIdx.x >> 5;
    const int nTilesN = DN / 32;                    // 16
    const int mblk = blockIdx.x / nTilesN;
    const int nblk = blockIdx.x % nTilesN;
    const int row0 = mblk * 256 + wave * 32;        // wave owns rows [row0, row0+32)
    const int col0 = nblk * 32;
    const int l15 = lane & 15;
    const int lh  = lane >> 4;
    const size_t G = (size_t)DN * DK;               // per-gate stride

    // A fragments (two M-tiles each for x and h)
    const unsigned short* xr0 = xb + (size_t)(row0 + l15) * DK + lh * 16;
    const unsigned short* xr1 = xr0 + (size_t)16 * DK;
    const unsigned short* hr0 = hb + (size_t)(row0 + l15) * DK + lh * 16;
    const unsigned short* hr1 = hr0 + (size_t)16 * DK;
    // B fragments: lane holds column; 32 contiguous bytes per lane per K-step.
    const unsigned short* bx0 = WxT + (size_t)(col0 + l15) * DK + lh * 16;
    const unsigned short* bx1 = WxT + (size_t)(col0 + 16 + l15) * DK + lh * 16;
    const unsigned short* bh0 = WhT + (size_t)(col0 + l15) * DK + lh * 16;
    const unsigned short* bh1 = WhT + (size_t)(col0 + 16 + l15) * DK + lh * 16;

    // accumulators: [m-tile][n-tile]; u,r fold x*W and h*W into one acc.
    v8f aU00 = {}, aU01 = {}, aU10 = {}, aU11 = {};
    v8f aR00 = {}, aR01 = {}, aR10 = {}, aR11 = {};
    v8f aCx00 = {}, aCx01 = {}, aCx10 = {}, aCx11 = {};
    v8f aCh00 = {}, aCh01 = {}, aCh10 = {}, aCh11 = {};

    // software pipeline: A fragments for step k preloaded during step k-1
    Frag ax0, ax1, ah0, ah1;
    LOADA(ax0, xr0, 0);
    LOADA(ax1, xr1, 0);
    LOADA(ah0, hr0, 0);
    LOADA(ah1, hr1, 0);

    for (int k0 = 0; k0 < DK; k0 += 32) {
        Frag b0, b1, b2, b3, b4, b5, b6, b7, b8, b9, b10, b11;
        LOADB(b0,  bx0, k0);            // Wu_x  n0
        LOADB(b1,  bx1, k0);            // Wu_x  n1
        LOADB(b2,  bh0, k0);            // Wu_h  n0
        LOADB(b3,  bh1, k0);            // Wu_h  n1
        LOADB(b4,  bx0 + G, k0);        // Wr_x  n0
        LOADB(b5,  bx1 + G, k0);        // Wr_x  n1
        LOADB(b6,  bh0 + G, k0);        // Wr_h  n0
        LOADB(b7,  bh1 + G, k0);        // Wr_h  n1
        LOADB(b8,  bx0 + 2 * G, k0);    // Wc_x  n0
        LOADB(b9,  bx1 + 2 * G, k0);    // Wc_x  n1
        LOADB(b10, bh0 + 2 * G, k0);    // Wc_h  n0
        LOADB(b11, bh1 + 2 * G, k0);    // Wc_h  n1

        // next step's A fragments (k wraps on last iter; harmless reload)
        const int kn = (k0 + 32) & (DK - 1);
        Frag axn0, axn1, ahn0, ahn1;
        LOADA(axn0, xr0, kn);
        LOADA(axn1, xr1, kn);
        LOADA(ahn0, hr0, kn);
        LOADA(ahn1, hr1, kn);

        // 24 WMMAs; each B fragment feeds both M-tiles back-to-back
        MMA(aU00,  ax0, b0);  MMA(aU10,  ax1, b0);
        MMA(aU01,  ax0, b1);  MMA(aU11,  ax1, b1);
        MMA(aU00,  ah0, b2);  MMA(aU10,  ah1, b2);
        MMA(aU01,  ah0, b3);  MMA(aU11,  ah1, b3);
        MMA(aR00,  ax0, b4);  MMA(aR10,  ax1, b4);
        MMA(aR01,  ax0, b5);  MMA(aR11,  ax1, b5);
        MMA(aR00,  ah0, b6);  MMA(aR10,  ah1, b6);
        MMA(aR01,  ah0, b7);  MMA(aR11,  ah1, b7);
        MMA(aCx00, ax0, b8);  MMA(aCx10, ax1, b8);
        MMA(aCx01, ax0, b9);  MMA(aCx11, ax1, b9);
        MMA(aCh00, ah0, b10); MMA(aCh10, ah1, b10);
        MMA(aCh01, ah0, b11); MMA(aCh11, ah1, b11);

        ax0 = axn0; ax1 = axn1; ah0 = ahn0; ah1 = ahn1;
    }

    // Epilogue. C layout: VGPR v, lanes0-15 -> M=v; lanes16-31 -> M=v+8; N=lane&15.
    const int colA = col0 + l15;
    const int colB = col0 + 16 + l15;
    const float buA = bu[colA], buB = bu[colB];
    const float brA = br[colA], brB = br[colB];
    const float bcA = bc[colA], bcB = bc[colB];

#define EPILOG(aU, aR, aCx, aCh, mt, col, bu_, br_, bc_)                        \
    {                                                                           \
        _Pragma("unroll")                                                       \
        for (int v = 0; v < 8; ++v) {                                           \
            const int row = row0 + (mt) * 16 + v + lh * 8;                      \
            const float a = att[row];                                           \
            const size_t base = (size_t)row * DN + (col);                       \
            float u = sigmoidf_(aU[v] + (bu_));                                 \
            float r = sigmoidf_(aR[v] + (br_));                                 \
            float c = tanhf_(aCx[v] + (bc_) + r * aCh[v]);                      \
            float s = state[base];                                              \
            float up = a * u;                                                   \
            out[base] = (1.0f - up) * s + up * c;                               \
        }                                                                       \
    }

    EPILOG(aU00, aR00, aCx00, aCh00, 0, colA, buA, brA, bcA)
    EPILOG(aU01, aR01, aCx01, aCh01, 0, colB, buB, brB, bcB)
    EPILOG(aU10, aR10, aCx10, aCh10, 1, colA, buA, brA, bcA)
    EPILOG(aU11, aR11, aCx11, aCh11, 1, colB, buB, brB, bcB)
#undef EPILOG
}

extern "C" void kernel_launch(void* const* d_in, const int* in_sizes, int n_in,
                              void* d_out, int out_size, void* d_ws, size_t ws_size,
                              hipStream_t stream) {
    const float* x    = (const float*)d_in[0];
    const float* h    = (const float*)d_in[1];
    const float* att  = (const float*)d_in[2];
    const float* Wu_x = (const float*)d_in[3];
    const float* bu   = (const float*)d_in[4];
    const float* Wu_h = (const float*)d_in[5];
    const float* Wr_x = (const float*)d_in[6];
    const float* br   = (const float*)d_in[7];
    const float* Wr_h = (const float*)d_in[8];
    const float* Wc_x = (const float*)d_in[9];
    const float* bc   = (const float*)d_in[10];
    const float* Wc_h = (const float*)d_in[11];
    float* out = (float*)d_out;

    const size_t nXH = (size_t)BATCH * DK;           // 16,777,216
    unsigned short* xb  = (unsigned short*)d_ws;
    unsigned short* hb  = xb + nXH;
    unsigned short* WxT = hb + nXH;
    unsigned short* WhT = WxT + (size_t)3 * DN * DK; // total ws: ~67 MB

    // bf16 conversion of activations (8 elems/thread)
    const int cvtBlocks = (int)(nXH / 8 / 256);      // 8192
    cvt_bf16x8<<<cvtBlocks, 256, 0, stream>>>(x, xb, (int)nXH);
    cvt_bf16x8<<<cvtBlocks, 256, 0, stream>>>(h, hb, (int)nXH);

    // weight convert+transpose: 6 matrices of 512x512
    wtrans<<<dim3(DK / 32, DN / 32, 6), dim3(32, 8), 0, stream>>>(
        Wu_x, Wr_x, Wc_x, Wu_h, Wr_h, Wc_h, WxT, WhT);

    // fused AUGRU: (32768/256) * (512/32) = 2048 blocks of 256 threads
    augru_wmma<<<(BATCH / 256) * (DN / 32), 256, 0, stream>>>(
        xb, hb, WxT, WhT, bu, br, bc, att, h, out);
}
